// SpatialEmbeddingLoss_30657476559542
// MI455X (gfx1250) — compile-verified, matrix operand order checked by reading, not run on hardware
//
#include <hip/hip_runtime.h>

// ---------------------------------------------------------------------------
// SpatialEmbeddingLoss for MI455X (gfx1250, wave32, WMMA + TDM).
//
// Pipeline (all on `stream`, graph-capture safe):
//   1) zeroWs        : clear stats/weights/meta/histograms in d_ws
//   2) phaseA        : tanh/sigmoid transforms + per-instance masked moments
//                      (LDS float atomics -> global atomics), seed_bg
//   3) finalizeStats : per-(b,k) center / sig_mean / smooth loss / quadratic
//                      GEMM weights  w_k = [-s, 2sc, -Σsc², 0]
//   4) wmmaProbs     : TDM loads the 16x8 weight matrix into LDS
//                      (tensor_load_to_lds + s_wait_tensorcnt), then N/16
//                      tiles x 16 instances via v_wmma_f32_16x16x32_f16,
//                      2 tiles in flight to hide WMMA->VALU hazards.
//                      exponent = f·w_k ; probs = exp(·); error histograms
//                      (4096 bins, errors ∈ [0,2]) + masked seed loss
//   5) lovaszFromHist: closed-form Lovasz hinge from descending histogram
//   6) finalLoss     : combine -> single f32 scalar
// ---------------------------------------------------------------------------

typedef __attribute__((ext_vector_type(16))) _Float16 v16h;
typedef __attribute__((ext_vector_type(8)))  float    v8f;
typedef __attribute__((ext_vector_type(4)))  unsigned int u32x4;
typedef __attribute__((ext_vector_type(8)))  int          i32x8;
typedef __attribute__((ext_vector_type(4)))  int          i32x4;

// therock-10.0 headers (amdgpu-toolchain / clang-23) ship the TDM header and
// pair with the 6-arg tensor_load_to_lds builtin; ROCm 7.2 uses the 5-arg one.
#if __has_include(<hip/amd_detail/amd_gfx1250_TDM.h>)
#define SEG_TDM_6ARG 1
#else
#define SEG_TDM_6ARG 0
#endif

namespace seg {
constexpr int Dd = 96, Hh = 96, Ww = 96;
constexpr int N  = Dd * Hh * Ww;        // 884736
constexpr int B  = 2;
constexpr int K  = 16;                  // instances 1..16
constexpr int NB = 4096;                // histogram bins over err in [0,2]
constexpr float VOX = 0.3f;

// workspace layout (float offsets)
constexpr size_t STATS_OFF = 0;                          // B*K*16 (cnt, Σe[3], Σσ[3], Σσ²[3])
constexpr size_t STATS_SZ  = (size_t)B * K * 16;         // 512
constexpr size_t WGT_OFF   = STATS_OFF + STATS_SZ;       // B*K*8 GEMM weights
constexpr size_t WGT_SZ    = (size_t)B * K * 8;          // 256
constexpr size_t META_OFF  = WGT_OFF + WGT_SZ;           // per seg: present, smooth, seed, lovasz
constexpr size_t META_SZ   = (size_t)B * K * 4 + B + 30; // + per-batch seed_bg, padded
constexpr size_t HIST_OFF  = META_OFF + META_SZ;         // B*K*NB*4 (fgCnt,fgSum,bgCnt,bgSum)
constexpr size_t HIST_SZ   = (size_t)B * K * NB * 4;     // 524288
constexpr size_t WS_END    = HIST_OFF + HIST_SZ;         // ~2.1 MB total
constexpr size_t SEEDBG_OFF = META_OFF + (size_t)B * K * 4;

constexpr int TILES_PER_BLOCK = 16;     // 16 WMMA tiles (256 voxels) per wave
} // namespace seg

using namespace seg;

__device__ __forceinline__ float sigmoidf_(float x) {
  return __builtin_amdgcn_rcpf(1.f + __expf(-x));   // v_rcp_f32, no precise-div
}

// ---------------------------------------------------------------- 1) zero ws
__global__ void zeroWs(float* ws) {
  size_t i = (size_t)blockIdx.x * blockDim.x + threadIdx.x;
  if (i < WS_END) ws[i] = 0.f;
}

// ------------------------------------------------- 2) transforms + moments
__global__ void phaseA(const float* __restrict__ off, const float* __restrict__ sig,
                       const float* __restrict__ seedIn, const int* __restrict__ tgt,
                       float* __restrict__ ws) {
  __shared__ float red[K * 10 + 1];
  const int tid = threadIdx.x;
  const int b   = blockIdx.y;
  const int v   = blockIdx.x * blockDim.x + tid;   // N divisible by 256
  for (int i = tid; i < K * 10 + 1; i += blockDim.x) red[i] = 0.f;
  __syncthreads();

  const int  w = v % Ww, h = (v / Ww) % Hh, d = v / (Ww * Hh);
  const float sx = (Ww * VOX) / (Ww - 1), sy = (Hh * VOX) / (Hh - 1), sz = (Dd * VOX) / (Dd - 1);
  const size_t base = (size_t)b * 3 * N + v;
  const float ex = tanhf(off[base])                 + w * sx;
  const float ey = tanhf(off[base + N])             + h * sy;
  const float ez = tanhf(off[base + 2 * (size_t)N]) + d * sz;
  const float s0 = sig[base], s1 = sig[base + N], s2 = sig[base + 2 * (size_t)N];
  const float sm = sigmoidf_(seedIn[(size_t)b * N + v]);
  const int   t  = tgt[(size_t)b * N + v];

  if (t == 0) {
    atomicAdd(&red[K * 10], sm * sm);                    // seed_bg
  } else {
    float* r = &red[(t - 1) * 10];
    atomicAdd(&r[0], 1.f);
    atomicAdd(&r[1], ex); atomicAdd(&r[2], ey); atomicAdd(&r[3], ez);
    atomicAdd(&r[4], s0); atomicAdd(&r[5], s1); atomicAdd(&r[6], s2);
    atomicAdd(&r[7], s0 * s0); atomicAdd(&r[8], s1 * s1); atomicAdd(&r[9], s2 * s2);
  }
  __syncthreads();
  if (tid < K * 10) {
    const int k = tid / 10, j = tid % 10;
    if (red[tid] != 0.f)
      atomicAdd(&ws[STATS_OFF + ((size_t)b * K + k) * 16 + j], red[tid]);
  }
  if (tid == K * 10 && red[K * 10] != 0.f)
    atomicAdd(&ws[SEEDBG_OFF + b], red[K * 10]);
}

// ---------------------------------------- 3) per-instance stats -> weights
__global__ void finalizeStats(float* __restrict__ ws) {
  const int t = threadIdx.x;
  if (t >= B * K) return;
  const float* st = ws + STATS_OFF + (size_t)t * 16;
  const float cnt = st[0];
  const float present = cnt > 0.f ? 1.f : 0.f;
  const float denom = fmaxf(cnt, 1.f);
  const float c0 = st[1] / denom, c1 = st[2] / denom, c2 = st[3] / denom;
  const float m0 = st[4] / denom, m1 = st[5] / denom, m2 = st[6] / denom;
  const float smooth =
      ((st[7] - 2.f * m0 * st[4] + m0 * m0 * cnt) +
       (st[8] - 2.f * m1 * st[5] + m1 * m1 * cnt) +
       (st[9] - 2.f * m2 * st[6] + m2 * m2 * cnt)) / (3.f * denom);
  const float e0 = __expf(10.f * m0), e1 = __expf(10.f * m1), e2 = __expf(10.f * m2);
  float* wv = ws + WGT_OFF + (size_t)t * 8;
  wv[0] = -e0;           wv[1] = -e1;           wv[2] = -e2;
  wv[3] = 2.f * e0 * c0; wv[4] = 2.f * e1 * c1; wv[5] = 2.f * e2 * c2;
  wv[6] = -(e0 * c0 * c0 + e1 * c1 * c1 + e2 * c2 * c2);
  wv[7] = 0.f;
  float* meta = ws + META_OFF + (size_t)t * 4;
  meta[0] = present;
  meta[1] = smooth;
}

// ------------------------- 4) Gaussian fields via WMMA + error histograms

// Build A fragment (16x32 f16): lanes 0-15 hold row M=lane&15, halves 0..7 =
// features K=0..7 = [ex²,ey²,ez²,ex,ey,ez,1,0]; lanes 16-31 (K=8..15/24..31)
// stay zero.
__device__ __forceinline__ v16h makeA(const float* __restrict__ offB, int vbase,
                                      int lo, int hi) {
  v16h af = {};
  const int v = vbase + lo;
  __builtin_prefetch(&offB[(v + 512 < N) ? v + 512 : v], 0, 1);   // global_prefetch_b8
  const int  w = v % Ww, h = (v / Ww) % Hh, d = v / (Ww * Hh);
  const float sx = (Ww * VOX) / (Ww - 1), sy = (Hh * VOX) / (Hh - 1), sz = (Dd * VOX) / (Dd - 1);
  const float ex = tanhf(offB[v])                 + w * sx;
  const float ey = tanhf(offB[v + N])             + h * sy;
  const float ez = tanhf(offB[v + 2 * (size_t)N]) + d * sz;
  const float f[8] = {ex * ex, ey * ey, ez * ez, ex, ey, ez, 1.f, 0.f};
  if (hi == 0) {
#pragma unroll
    for (int hh = 0; hh < 8; ++hh) af[hh] = (_Float16)f[hh];
  }
  return af;
}

// Consume one 16x16 D tile: probs, Lovasz error histogram, masked seed loss.
// C/D layout: reg r, lanes 0-15 -> M=r ; lanes 16-31 -> M=8+r ; N = lane&15.
__device__ __forceinline__ void emitTile(const v8f& acc, int vbase, int hi, int kInst,
                                         const int* __restrict__ tgB,
                                         const float* __restrict__ seedB,
                                         float* __restrict__ hseg0,
                                         float* __restrict__ seedAcc) {
#pragma unroll
  for (int r = 0; r < 8; ++r) {
    const int   v   = vbase + r + 8 * hi;
    const int   t   = tgB[v];
    const bool  fg  = (t == kInst + 1);
    const float p   = __expf(fminf(acc[r], 0.f));       // exponent <= 0 -> p in (0,1]
    const float err = fg ? (2.f - 2.f * p) : (2.f * p); // Lovasz hinge error, in [0,2]
    int bin = (int)(err * (NB * 0.5f));
    bin = bin < 0 ? 0 : (bin > NB - 1 ? NB - 1 : bin);
    float* hseg = hseg0 + (size_t)bin * 4;
    if (fg) {
      atomicAdd(hseg + 0, 1.f);
      atomicAdd(hseg + 1, err);
      const float sv = sigmoidf_(seedB[v]);
      atomicAdd(seedAcc, (sv - p) * (sv - p));          // W_FG = 1
    } else {
      atomicAdd(hseg + 2, 1.f);
      atomicAdd(hseg + 3, err);
    }
  }
}

// One wave (32 threads) per block. TDM pulls the per-batch 16x8 f32 weight
// matrix into LDS; each wave then runs TILES_PER_BLOCK WMMAs, 2 in flight.
__global__ void __launch_bounds__(32)
wmmaProbs(const float* __restrict__ off, const float* __restrict__ seedIn,
          const int* __restrict__ tgt, float* __restrict__ ws) {
  __shared__ float lw[K * 8];           // only LDS allocation -> offset 0
  const int lane  = threadIdx.x;        // 0..31, wave32
  const int b     = blockIdx.y;
  const int tile0 = blockIdx.x * TILES_PER_BLOCK;
  const int hi    = lane >> 4;          // half-wave id
  const int lo    = lane & 15;
  const int kInst = lo;                 // output instance column (C/D layout: N = lane&15)

  const float* offB  = off + (size_t)b * 3 * N;
  const float* seedB = seedIn + (size_t)b * N;
  const int*   tgB   = tgt + (size_t)b * N;

  // --- TDM: 1-D tensor, 128 f32 elements (16 instances x 8 weights) -> LDS.
  // D# group0: count=1, lds_addr=0, global_addr, type=2 (bits 127:126).
  // D# group1: data_size=4B (bits 17:16), tensor_dim0=128 (bit 48),
  //            tensor_dim1=1 (bit 80), tile_dim0=128 (bit 112),
  //            tile_dim1=1 (bit 128), tensor_dim0_stride=128 (bit 160).
  {
    const unsigned long long ga =
        (unsigned long long)(uintptr_t)(ws + WGT_OFF + (size_t)b * K * 8);
    u32x4 g0 = {0u, 0u, 0u, 0u};
    g0[0] = 1u;                                    // count=1 (user descriptor)
    g0[1] = 0u;                                    // lds_addr = offset of lw
    g0[2] = (unsigned)(ga & 0xffffffffull);        // global_addr[31:0]
    g0[3] = (unsigned)((ga >> 32) & 0x01ffffffull) | (2u << 30);  // [56:32] | type=2
    i32x8 g1 = {0, 0, 0, 0, 0, 0, 0, 0};
    g1[0] = 0x00020000;                            // data_size = 4 bytes
    g1[1] = (int)(128u << 16);                     // tensor_dim0 = 128
    g1[2] = (int)(1u << 16);                       // tensor_dim1 = 1
    g1[3] = (int)(128u << 16);                     // tile_dim0 = 128
    g1[4] = 1;                                     // tile_dim1 = 1
    g1[5] = 128;                                   // tensor_dim0_stride = 128
    i32x4 z4 = {0, 0, 0, 0};
#if SEG_TDM_6ARG
    i32x8 z8 = {0, 0, 0, 0, 0, 0, 0, 0};
    __builtin_amdgcn_tensor_load_to_lds(g0, g1, z4, z4, z8, 0);
#else
    __builtin_amdgcn_tensor_load_to_lds(g0, g1, z4, z4, 0);
#endif
    __builtin_amdgcn_s_wait_tensorcnt(0);          // s_wait_tensorcnt 0x0
  }

  // B fragment (32x16 f16) from LDS: lane L holds column N=L&15; half h ->
  // K = h + 8*(L>>4) (h<8), 8 + h + 8*(L>>4) (h>=8). Only K=0..7 nonzero -> lanes 0-15, halves 0..7.
  v16h bf = {};
  if (hi == 0) {
#pragma unroll
    for (int hh = 0; hh < 8; ++hh) bf[hh] = (_Float16)lw[lo * 8 + hh];
  }

  float* hseg0   = ws + HIST_OFF + ((size_t)b * K + kInst) * NB * 4;
  float* seedAcc = ws + META_OFF + ((size_t)b * K + kInst) * 4 + 2;

  for (int ti = 0; ti < TILES_PER_BLOCK; ti += 2) {
    const int vb0 = (tile0 + ti) * 16;
    const int vb1 = vb0 + 16;
    const v16h a0 = makeA(offB, vb0, lo, hi);
    const v16h a1 = makeA(offB, vb1, lo, hi);
    v8f acc0 = {}, acc1 = {};
    acc0 = __builtin_amdgcn_wmma_f32_16x16x32_f16(false, a0, false, bf,
                                                  (short)0, acc0, false, false);
    acc1 = __builtin_amdgcn_wmma_f32_16x16x32_f16(false, a1, false, bf,
                                                  (short)0, acc1, false, false);
    emitTile(acc0, vb0, hi, kInst, tgB, seedB, hseg0, seedAcc);
    emitTile(acc1, vb1, hi, kInst, tgB, seedB, hseg0, seedAcc);
  }
}

// ----------------------- 5) Lovasz hinge from descending error histogram
// fg elems: each contributes err/(G+Bg)  (exact, union constant across fg run)
// bg elems: gradient telescopes ->  avg_err * I * (1/U0 - 1/U1) per bin
__global__ void lovaszFromHist(float* __restrict__ ws) {
  const int s = blockIdx.x;             // 0..B*K-1
  if (threadIdx.x != 0) return;
  const float G = ws[STATS_OFF + (size_t)s * 16];   // fg count
  float loss = 0.f;
  if (G > 0.5f) {
    const float* hseg = ws + HIST_OFF + (size_t)s * NB * 4;
    float F = 0.f, Bg = 0.f;
    for (int bin = NB - 1; bin >= 0; --bin) {
      const float nf = hseg[bin * 4 + 0], sf = hseg[bin * 4 + 1];
      const float nb = hseg[bin * 4 + 2], sb = hseg[bin * 4 + 3];
      if (nf > 0.f) { loss += sf / (G + Bg); F += nf; }
      if (nb > 0.f) {
        const float I = G - F;
        const float U0 = G + Bg, U1 = U0 + nb;
        loss += (sb / nb) * I * (1.f / U0 - 1.f / U1);
        Bg += nb;
      }
    }
  }
  ws[META_OFF + (size_t)s * 4 + 3] = loss;
}

// ------------------------------------------------------------- 6) combine
__global__ void finalLoss(const float* __restrict__ ws, float* __restrict__ out) {
  if (threadIdx.x != 0 || blockIdx.x != 0) return;
  float total = 0.f;
  for (int b = 0; b < B; ++b) {
    float obj = 0.f, inst = 0.f, smooth = 0.f, seedL = 0.f;
    for (int k = 0; k < K; ++k) {
      const float* m = ws + META_OFF + ((size_t)b * K + k) * 4;
      obj    += m[0];
      inst   += m[0] * m[3];
      smooth += m[0] * m[1];
      seedL  += m[0] * m[2];
    }
    const float safe = fmaxf(obj, 1.f);
    const float sbg  = ws[SEEDBG_OFF + b];
    total += inst / safe + smooth / safe + 10.f * (sbg + seedL) / (float)N;
  }
  out[0] = total / (float)B;
}

// ---------------------------------------------------------------------------
extern "C" void kernel_launch(void* const* d_in, const int* in_sizes, int n_in,
                              void* d_out, int out_size, void* d_ws, size_t ws_size,
                              hipStream_t stream) {
  (void)in_sizes; (void)n_in; (void)out_size; (void)ws_size;
  const float* off    = (const float*)d_in[0];   // [B,3,D,H,W]
  const float* sig    = (const float*)d_in[1];   // [B,3,D,H,W]
  const float* seedIn = (const float*)d_in[2];   // [B,1,D,H,W]
  const int*   tgt    = (const int*)d_in[3];     // [B,D,H,W]
  float* ws  = (float*)d_ws;
  float* out = (float*)d_out;

  zeroWs<<<dim3((unsigned)((WS_END + 255) / 256)), dim3(256), 0, stream>>>(ws);
  phaseA<<<dim3(N / 256, B), dim3(256), 0, stream>>>(off, sig, seedIn, tgt, ws);
  finalizeStats<<<dim3(1), dim3(32), 0, stream>>>(ws);
  wmmaProbs<<<dim3((N / 16) / TILES_PER_BLOCK, B), dim3(32), 0, stream>>>(off, seedIn, tgt, ws);
  lovaszFromHist<<<dim3(B * K), dim3(32), 0, stream>>>(ws);
  finalLoss<<<dim3(1), dim3(32), 0, stream>>>(ws, out);
}
